// DeformableInceptionModule_51238959841383
// MI455X (gfx1250) — matrix operand order repaired and checked by reading, not compiled, not run on hardware
//
#include <hip/hip_runtime.h>

typedef __attribute__((ext_vector_type(16))) _Float16 v16h;
typedef __attribute__((ext_vector_type(8)))  float    v8f;

#define HW        4096
#define Himg      64
#define Wimg      64
#define Cin       64
#define CoutC     64
#define CoutTotal 192
#define MTILE     128
#define APITCH    72   // halves
#define BPITCH    72   // halves
#define CPITCH    132  // floats

// Transpose + convert weights: wT[k][co][c] = (f16) w[co][c][k]
__global__ void prep_weights_kernel(const float* __restrict__ w,
                                    _Float16* __restrict__ wT, int K) {
  int idx = blockIdx.x * blockDim.x + threadIdx.x;
  int total = K * CoutC * Cin;
  if (idx >= total) return;
  int c  = idx % Cin;
  int co = (idx / Cin) % CoutC;
  int k  = idx / (Cin * CoutC);
  wT[idx] = (_Float16)w[(co * Cin + c) * K + k];
}

template<int KH, int KW>
__global__ __launch_bounds__(256) void dcn_wmma_kernel(
    const float* __restrict__ x,
    const _Float16* __restrict__ wT,
    const float* __restrict__ offset,
    const float* __restrict__ mask,
    float* __restrict__ out,
    int coBase)
{
  constexpr int K   = KH * KW;
  constexpr int PAD = (KH - 1) / 2;
  constexpr int ABYTES = MTILE * APITCH * 2;   // 18432
  constexpr int BBYTES = CoutC * BPITCH * 2;   //  9216

  // Double-buffered A/B tiles; C staging aliases the A buffers (disjoint lifetime).
  __shared__ __align__(16) unsigned char smem[2 * (ABYTES + BBYTES)];  // 54 KB
  float* Cs = (float*)smem;                    // 33792 B < 2*ABYTES

  const int tid  = threadIdx.x;
  const int wave = tid >> 5;
  const int lane = tid & 31;
  const int r    = lane & 15;
  const int hi   = lane >> 4;

  const int b      = blockIdx.x >> 5;          // 32 tiles per image
  const int hwBase = (blockIdx.x & 31) * MTILE;

  const int m0 = (wave & 3) * 32;              // wave's 32x32 output block
  const int n0 = (wave >> 2) * 32;

  v8f acc00 = {}, acc01 = {}, acc10 = {}, acc11 = {};

  // sampling assignment: 2 threads per spatial position, 32 channels each
  const int pos   = tid >> 1;                  // 0..127
  const int chalf = (tid & 1) * 32;
  const int hw = hwBase + pos;
  const int h  = hw >> 6;
  const int w_ = hw & 63;
  const float* xb    = x      + (size_t)b * Cin * HW;
  const float* offb  = offset + (size_t)b * 2 * K * HW + hw;
  const float* maskb = mask   + (size_t)b * K * HW + hw;

  // Fill buffer `buf` with tap k's A tile (bilinear samples) and B tile (weights)
  auto sampleTap = [&](int k, int buf) {
    _Float16* As = (_Float16*)(smem + (buf ? ABYTES : 0));
    _Float16* Bs = (_Float16*)(smem + 2 * ABYTES + (buf ? BBYTES : 0));
    // B tile: coalesced 32B per thread
    {
      int co  = tid >> 2;
      int seg = tid & 3;
      const uint4* src = (const uint4*)(wT + ((size_t)k * CoutC + co) * Cin + seg * 16);
      uint4* dst = (uint4*)(Bs + co * BPITCH + seg * 16);
      dst[0] = src[0];
      dst[1] = src[1];
    }
    // A tile: gather + modulate + cvt f16
    const int ky = k / KW, kx = k % KW;
    float dy = offb[(size_t)(2 * k) * HW];
    float dx = offb[(size_t)(2 * k + 1) * HW];
    float mk = maskb[(size_t)k * HW];
    float py = (float)(h - PAD + ky) + dy;
    float px = (float)(w_ - PAD + kx) + dx;
    float y0f = floorf(py), x0f = floorf(px);
    float wy = py - y0f, wx = px - x0f;
    int y0 = (int)y0f, x0 = (int)x0f;
    int y1 = y0 + 1,   x1 = x0 + 1;
    bool vy0 = (y0 >= 0) && (y0 < Himg);
    bool vy1 = (y1 >= 0) && (y1 < Himg);
    bool vx0 = (x0 >= 0) && (x0 < Wimg);
    bool vx1 = (x1 >= 0) && (x1 < Wimg);
    int yc0 = min(max(y0, 0), Himg - 1);
    int yc1 = min(max(y1, 0), Himg - 1);
    int xc0 = min(max(x0, 0), Wimg - 1);
    int xc1 = min(max(x1, 0), Wimg - 1);
    float c00 = (vy0 && vx0) ? (1.f - wy) * (1.f - wx) * mk : 0.f;
    float c01 = (vy0 && vx1) ? (1.f - wy) * wx * mk : 0.f;
    float c10 = (vy1 && vx0) ? wy * (1.f - wx) * mk : 0.f;
    float c11 = (vy1 && vx1) ? wy * wx * mk : 0.f;
    int i00 = yc0 * Wimg + xc0;
    int i01 = yc0 * Wimg + xc1;
    int i10 = yc1 * Wimg + xc0;
    int i11 = yc1 * Wimg + xc1;
    const float* xc = xb + (size_t)chalf * HW;
    _Float16* arow = As + pos * APITCH + chalf;
    #pragma unroll 4
    for (int c = 0; c < 32; ++c) {
      const float* p = xc + (size_t)c * HW;
      float s = c00 * p[i00] + c01 * p[i01] + c10 * p[i10] + c11 * p[i11];
      arow[c] = (_Float16)s;
    }
  };

  sampleTap(0, 0);
  __syncthreads();

  #pragma unroll 2
  for (int k = 0; k < K; ++k) {
    const int cur = k & 1;
    const _Float16* As = (const _Float16*)(smem + (cur ? ABYTES : 0));
    const _Float16* Bs = (const _Float16*)(smem + 2 * ABYTES + (cur ? BBYTES : 0));

    // ---- consume tap k: 2 x (16x16x32) K-steps, 2x2 fragment block ----
    #pragma unroll
    for (int ks = 0; ks < 64; ks += 32) {
      v16h a0, a1, bv0, bv1;
      {
        const _Float16* row = As + (m0 + r) * APITCH + ks + 8 * hi;
        ((uint4*)&a0)[0] = *(const uint4*)(row);
        ((uint4*)&a0)[1] = *(const uint4*)(row + 16);
        const _Float16* row1 = row + 16 * APITCH;
        ((uint4*)&a1)[0] = *(const uint4*)(row1);
        ((uint4*)&a1)[1] = *(const uint4*)(row1 + 16);
      }
      {
        const _Float16* rowb = Bs + (n0 + r) * BPITCH + ks + 8 * hi;
        ((uint4*)&bv0)[0] = *(const uint4*)(rowb);
        ((uint4*)&bv0)[1] = *(const uint4*)(rowb + 16);
        const _Float16* rowb1 = rowb + 16 * BPITCH;
        ((uint4*)&bv1)[0] = *(const uint4*)(rowb1);
        ((uint4*)&bv1)[1] = *(const uint4*)(rowb1 + 16);
      }
      acc00 = __builtin_amdgcn_wmma_f32_16x16x32_f16(false, a0, false, bv0, (short)0, acc00, false, false);
      acc01 = __builtin_amdgcn_wmma_f32_16x16x32_f16(false, a0, false, bv1, (short)0, acc01, false, false);
      acc10 = __builtin_amdgcn_wmma_f32_16x16x32_f16(false, a1, false, bv0, (short)0, acc10, false, false);
      acc11 = __builtin_amdgcn_wmma_f32_16x16x32_f16(false, a1, false, bv1, (short)0, acc11, false, false);
    }

    // ---- overlap: prefetch+sample tap k+1 into the other buffer ----
    if (k + 1 < K) sampleTap(k + 1, cur ^ 1);

    __syncthreads();
  }

  // ---- stage accumulators to LDS transposed: Cs[n][m], m contiguous ----
  #pragma unroll
  for (int e = 0; e < 8; ++e) {
    int mm = e + 8 * hi;
    Cs[(n0 + r)      * CPITCH + m0 + mm]      = acc00[e];
    Cs[(n0 + 16 + r) * CPITCH + m0 + mm]      = acc01[e];
    Cs[(n0 + r)      * CPITCH + m0 + 16 + mm] = acc10[e];
    Cs[(n0 + 16 + r) * CPITCH + m0 + 16 + mm] = acc11[e];
  }
  __syncthreads();

  // ---- coalesced output: out[b][coBase+co][hwBase + m] ----
  {
    int co  = tid >> 2;
    int seg = tid & 3;
    const float* src = Cs + co * CPITCH + seg * 32;
    float* dst = out + ((size_t)b * CoutTotal + coBase + co) * HW + hwBase + seg * 32;
    #pragma unroll
    for (int q = 0; q < 8; ++q)
      ((float4*)dst)[q] = ((const float4*)src)[q];
  }
}

extern "C" void kernel_launch(void* const* d_in, const int* in_sizes, int n_in,
                              void* d_out, int out_size, void* d_ws, size_t ws_size,
                              hipStream_t stream) {
  (void)in_sizes; (void)n_in; (void)out_size; (void)ws_size;
  const float* x   = (const float*)d_in[0];
  const float* w1  = (const float*)d_in[1];
  const float* of1 = (const float*)d_in[2];
  const float* mk1 = (const float*)d_in[3];
  const float* w2  = (const float*)d_in[4];
  const float* of2 = (const float*)d_in[5];
  const float* mk2 = (const float*)d_in[6];
  const float* w3  = (const float*)d_in[7];
  const float* of3 = (const float*)d_in[8];
  const float* mk3 = (const float*)d_in[9];
  float* out = (float*)d_out;

  _Float16* wT1 = (_Float16*)d_ws;
  _Float16* wT2 = wT1 + 9  * CoutC * Cin;
  _Float16* wT3 = wT2 + 25 * CoutC * Cin;

  prep_weights_kernel<<<(9  * CoutC * Cin + 255) / 256, 256, 0, stream>>>(w1, wT1, 9);
  prep_weights_kernel<<<(25 * CoutC * Cin + 255) / 256, 256, 0, stream>>>(w2, wT2, 25);
  prep_weights_kernel<<<(49 * CoutC * Cin + 255) / 256, 256, 0, stream>>>(w3, wT3, 49);

  dim3 grid(4 * HW / MTILE);   // 128 blocks
  dim3 block(256);
  dcn_wmma_kernel<3, 3><<<grid, block, 0, stream>>>(x, wT1, of1, mk1, out, 0);
  dcn_wmma_kernel<5, 5><<<grid, block, 0, stream>>>(x, wT2, of2, mk2, out, 64);
  dcn_wmma_kernel<7, 7><<<grid, block, 0, stream>>>(x, wT3, of3, mk3, out, 128);
}
